// SmoothQuantDynamic_43671227466064
// MI455X (gfx1250) — 8-bit matrix operand order repaired and checked by reading, compiled, not run on hardware
//
#include <hip/hip_runtime.h>
#include <stdint.h>


typedef int v8i __attribute__((ext_vector_type(8)));

#define T_DIM   8192
#define IN_DIM  4096
#define OUT_DIM 4096

__global__ void packw_kernel(const int* __restrict__ w, unsigned* __restrict__ w8, int n4) {
  const int stride = gridDim.x * blockDim.x;
  for (int i = blockIdx.x * blockDim.x + threadIdx.x; i < n4; i += stride) {
    int4 v = ((const int4*)w)[i];
    unsigned p = (unsigned)(v.x & 255) | ((unsigned)(v.y & 255) << 8) |
                 ((unsigned)(v.z & 255) << 16) | ((unsigned)(v.w & 255) << 24);
    w8[i] = p;
  }
}

__global__ void absmax_kernel(const float* __restrict__ x, const float* __restrict__ scale,
                              unsigned* __restrict__ out, int n4) {
  float m = 0.0f;
  const int stride = gridDim.x * blockDim.x;
  for (int i = blockIdx.x * blockDim.x + threadIdx.x; i < n4; i += stride) {
    float4 v = ((const float4*)x)[i];
    const int k = (i << 2) & (IN_DIM - 1);
    float4 s = *(const float4*)(scale + k);
    m = fmaxf(m, fabsf(v.x) / s.x);
    m = fmaxf(m, fabsf(v.y) / s.y);
    m = fmaxf(m, fabsf(v.z) / s.z);
    m = fmaxf(m, fabsf(v.w) / s.w);
  }
  #pragma unroll
  for (int off = 16; off > 0; off >>= 1)
    m = fmaxf(m, __shfl_xor(m, off, 32));
  if ((threadIdx.x & 31) == 0)
    atomicMax(out, __float_as_uint(m));
}

__global__ void quant_kernel(const float* __restrict__ x, const float* __restrict__ scale,
                             const unsigned* __restrict__ sxbits,
                             unsigned* __restrict__ xq, int n4) {
  const float r = 127.0f / __uint_as_float(*sxbits);
  const int stride = gridDim.x * blockDim.x;
  for (int i = blockIdx.x * blockDim.x + threadIdx.x; i < n4; i += stride) {
    float4 v = ((const float4*)x)[i];
    const int k = (i << 2) & (IN_DIM - 1);
    float4 s = *(const float4*)(scale + k);
    int q0 = (int)rintf(v.x / s.x * r);
    int q1 = (int)rintf(v.y / s.y * r);
    int q2 = (int)rintf(v.z / s.z * r);
    int q3 = (int)rintf(v.w / s.w * r);
    q0 = min(127, max(-127, q0));
    q1 = min(127, max(-127, q1));
    q2 = min(127, max(-127, q2));
    q3 = min(127, max(-127, q3));
    xq[i] = (unsigned)(q0 & 255) | ((unsigned)(q1 & 255) << 8) |
            ((unsigned)(q2 & 255) << 16) | ((unsigned)(q3 & 255) << 24);
  }
}

__global__ __launch_bounds__(256)
void gemm_kernel(const unsigned char* __restrict__ xq, const unsigned char* __restrict__ w8,
                 const unsigned* __restrict__ sxbits, const float* __restrict__ s_w,
                 const float* __restrict__ bias, float* __restrict__ out) {
  constexpr int LDA = 80;
  __shared__ alignas(16) unsigned char sA[128 * LDA];
  __shared__ alignas(16) unsigned char sB[128 * LDA];

  const int tid    = threadIdx.x;
  const int lane   = tid & 31;
  const int wave   = tid >> 5;
  const int laneM  = lane & 15;
  const int laneHi = lane >> 4;
  const int wm     = wave >> 2;
  const int wn     = wave & 3;

  const int m0 = blockIdx.y * 128;
  const int n0 = blockIdx.x * 128;

  v8i acc[4][2];
  #pragma unroll
  for (int mt = 0; mt < 4; ++mt)
    #pragma unroll
    for (int nt = 0; nt < 2; ++nt)
      #pragma unroll
      for (int j = 0; j < 8; ++j)
        acc[mt][nt][j] = 0;

  const int rowA = tid >> 1;
  const int half = (tid & 1) * 32;

  for (int k0 = 0; k0 < IN_DIM; k0 += 64) {
    {
      const uint4* gA = (const uint4*)(xq + (size_t)(m0 + rowA) * IN_DIM + k0 + half);
      uint4* dA = (uint4*)(sA + rowA * LDA + half);
      dA[0] = gA[0];
      dA[1] = gA[1];
      const uint4* gB = (const uint4*)(w8 + (size_t)(n0 + rowA) * IN_DIM + k0 + half);
      uint4* dB = (uint4*)(sB + rowA * LDA + half);
      dB[0] = gB[0];
      dB[1] = gB[1];
    }
    __syncthreads();

    v8i bfrag[2];
    #pragma unroll
    for (int nt = 0; nt < 2; ++nt) {
      const unsigned char* p = sB + (wn * 32 + nt * 16 + laneM) * LDA + laneHi * 16;
      uint4 b0 = *(const uint4*)(p);
      uint4 b1 = *(const uint4*)(p + 32);
      bfrag[nt][0] = (int)b0.x; bfrag[nt][1] = (int)b0.y;
      bfrag[nt][2] = (int)b0.z; bfrag[nt][3] = (int)b0.w;
      bfrag[nt][4] = (int)b1.x; bfrag[nt][5] = (int)b1.y;
      bfrag[nt][6] = (int)b1.z; bfrag[nt][7] = (int)b1.w;
    }

    #pragma unroll
    for (int mt = 0; mt < 4; ++mt) {
      const unsigned char* pa = sA + (wm * 64 + mt * 16 + laneM) * LDA + laneHi * 16;
      v8i afrag;
      #pragma unroll
      for (int j = 0; j < 8; ++j) {
        const int K = ((j >> 2) << 5) + ((j & 3) << 2);
        afrag[j] = *(const int*)(pa + K);
      }
      #pragma unroll
      for (int nt = 0; nt < 2; ++nt) {
        acc[mt][nt] = __builtin_amdgcn_wmma_i32_16x16x64_iu8(
             true, afrag,  true, bfrag[nt],
            acc[mt][nt],  false,  false);
      }
    }
    __syncthreads();
  }

  const float sx = __uint_as_float(*sxbits) * (1.0f / 127.0f);
  #pragma unroll
  for (int nt = 0; nt < 2; ++nt) {
    const int col = n0 + wn * 32 + nt * 16 + laneM;
    const float sc = sx * s_w[col];
    const float bb = bias[col];
    #pragma unroll
    for (int mt = 0; mt < 4; ++mt) {
      #pragma unroll
      for (int r = 0; r < 8; ++r) {
        const int row = m0 + wm * 64 + mt * 16 + laneHi * 8 + r;
        out[(size_t)row * OUT_DIM + col] = (float)acc[mt][nt][r] * sc + bb;
      }
    }
  }
}

extern "C" void kernel_launch(void* const* d_in, const int* in_sizes, int n_in,
                              void* d_out, int out_size, void* d_ws, size_t ws_size,
                              hipStream_t stream) {
  const float* x     = (const float*)d_in[0];
  const int*   w_q   = (const int*)d_in[1];
  const float* s_w   = (const float*)d_in[2];
  const float* scale = (const float*)d_in[3];
  const float* bias  = (const float*)d_in[4];
  float* out = (float*)d_out;

  unsigned char* ws = (unsigned char*)d_ws;
  unsigned* d_sx      = (unsigned*)ws;
  unsigned char* w8   = ws + 256;
  unsigned char* xq   = w8 + (size_t)OUT_DIM * IN_DIM;

  hipMemsetAsync(d_sx, 0, sizeof(unsigned), stream);

  const int wn4 = (OUT_DIM * IN_DIM) / 4;
  packw_kernel<<<2048, 256, 0, stream>>>(w_q, (unsigned*)w8, wn4);

  const int xn4 = (T_DIM * IN_DIM) / 4;
  absmax_kernel<<<2048, 256, 0, stream>>>(x, scale, d_sx, xn4);
  quant_kernel<<<4096, 256, 0, stream>>>(x, scale, d_sx, (unsigned*)xq, xn4);

  dim3 grid(OUT_DIM / 128, T_DIM / 128);
  gemm_kernel<<<grid, 256, 0, stream>>>(xq, w8, d_sx, s_w, bias, out);
}
